// Complex_Concat_Layer_59055800320843
// MI455X (gfx1250) — matrix operand order repaired and checked by reading, not compiled, not run on hardware
//
#include <hip/hip_runtime.h>
#include <hip/hip_bf16.h>

// Problem constants (from reference): B=2, C=8, L=1024, H=512
constexpr int Bc = 2;
constexpr int Cc = 8;
constexpr int Lc = 1024;
constexpr int Hc = 512;

constexpr int BM = 128;       // block tile rows (l)
constexpr int BN = 128;       // block tile cols (m)
constexpr int BK = 32;        // K chunk (one bf16 WMMA K)
constexpr int KCH = Hc / BK;  // 16 chunks
constexpr int LDSW = 40;      // padded LDS row stride in bf16 (80B -> conflict-free b128 reads)

typedef __attribute__((ext_vector_type(16))) __bf16 v16bf;
typedef __attribute__((ext_vector_type(4)))  __bf16 v4bf;
typedef __attribute__((ext_vector_type(8)))  float  v8f;

// ---------------------------------------------------------------------------
// Kernel 1: per-row dots  a[row] = S[row,:]·(v1+v3),  b[row] = E[row,:]·(v2-v3)
// One wave32 per row, 8 waves per block.
// ---------------------------------------------------------------------------
__global__ __launch_bounds__(256)
void prep_rowdots(const float* __restrict__ S, const float* __restrict__ E,
                  const float* __restrict__ v,
                  float* __restrict__ avec, float* __restrict__ bvec)
{
    const int wid  = threadIdx.x >> 5;
    const int lane = threadIdx.x & 31;
    const size_t row = (size_t)blockIdx.x * 8 + wid;   // 0 .. B*C*L-1
    const float* srow = S + row * Hc;
    const float* erow = E + row * Hc;

    float sa = 0.f, sb = 0.f;
    #pragma unroll
    for (int k = lane * 4; k < Hc; k += 32 * 4) {
        float4 s4 = *(const float4*)(srow + k);
        float4 e4 = *(const float4*)(erow + k);
        float4 v1 = *(const float4*)(v + k);
        float4 v2 = *(const float4*)(v + Hc + k);
        float4 v3 = *(const float4*)(v + 2 * Hc + k);
        sa += s4.x * (v1.x + v3.x) + s4.y * (v1.y + v3.y)
            + s4.z * (v1.z + v3.z) + s4.w * (v1.w + v3.w);
        sb += e4.x * (v2.x - v3.x) + e4.y * (v2.y - v3.y)
            + e4.z * (v2.z - v3.z) + e4.w * (v2.w - v3.w);
    }
    #pragma unroll
    for (int off = 16; off > 0; off >>= 1) {
        sa += __shfl_down(sa, off, 32);
        sb += __shfl_down(sb, off, 32);
    }
    if (lane == 0) { avec[row] = sa; bvec[row] = sb; }
}

// ---------------------------------------------------------------------------
// Kernel 2: per-(b,c) GEMM  M = (S*v4) @ E^T  via split-bf16 WMMA, +a[l]+b[m],
// stored to out[b, l, m, c].
// ---------------------------------------------------------------------------
__global__ __launch_bounds__(256)
void gemm_bclm(const float* __restrict__ S, const float* __restrict__ E,
               const float* __restrict__ v,
               const float* __restrict__ avec, const float* __restrict__ bvec,
               float* __restrict__ out)
{
    __shared__ __bf16 sAhi[BM * LDSW];
    __shared__ __bf16 sAlo[BM * LDSW];
    __shared__ __bf16 sBhi[BN * LDSW];
    __shared__ __bf16 sBlo[BN * LDSW];

    const int bc = blockIdx.z;            // b*C + c
    const int bb = bc / Cc;
    const int cc = bc % Cc;
    const int m0 = blockIdx.x * BN;       // m tile (E rows / output N)
    const int l0 = blockIdx.y * BM;       // l tile (S rows / output M)

    const int tid  = threadIdx.x;         // 0..255
    const int lane = tid & 31;
    const int wid  = tid >> 5;            // 0..7
    const int wm   = wid & 3;             // 4 waves along M (32 rows each)
    const int wn   = wid >> 2;            // 2 waves along N (64 cols each)

    const float* Sbase = S + (size_t)bc * Lc * Hc;
    const float* Ebase = E + (size_t)bc * Lc * Hc;
    const float* v4p   = v + 3 * Hc;

    // Staging: per chunk, each matrix is 128 rows x 8 float4 segs = 1024 float4
    // -> 4 per thread per matrix.  seg = tid%8, rows = tid/8 + {0,32,64,96}.
    float4 rs[4], re[4];
    const int seg = tid & 7;
    const int r0  = tid >> 3;

    auto load_chunk = [&](int kc) {
        const size_t koff = (size_t)kc * BK + seg * 4;
        #pragma unroll
        for (int i = 0; i < 4; ++i) {
            const int r = r0 + i * 32;
            rs[i] = *(const float4*)(Sbase + (size_t)(l0 + r) * Hc + koff);
            re[i] = *(const float4*)(Ebase + (size_t)(m0 + r) * Hc + koff);
        }
    };

    auto store_chunk = [&](int kc) {
        float4 v4v = *(const float4*)(v4p + kc * BK + seg * 4);
        #pragma unroll
        for (int i = 0; i < 4; ++i) {
            const int r = r0 + i * 32;
            float sx[4] = { rs[i].x * v4v.x, rs[i].y * v4v.y,
                            rs[i].z * v4v.z, rs[i].w * v4v.w };
            float ex[4] = { re[i].x, re[i].y, re[i].z, re[i].w };
            v4bf shi, slo, ehi, elo;
            #pragma unroll
            for (int j = 0; j < 4; ++j) {
                __bf16 h1 = (__bf16)sx[j];
                shi[j] = h1;
                slo[j] = (__bf16)(sx[j] - (float)h1);
                __bf16 h2 = (__bf16)ex[j];
                ehi[j] = h2;
                elo[j] = (__bf16)(ex[j] - (float)h2);
            }
            const int off = r * LDSW + seg * 4;     // 8-byte aligned
            *(v4bf*)(sAhi + off) = shi;
            *(v4bf*)(sAlo + off) = slo;
            *(v4bf*)(sBhi + off) = ehi;
            *(v4bf*)(sBlo + off) = elo;
        }
    };

    // Accumulators: 2 (M subtiles) x 4 (N subtiles) of 16x16 f32
    v8f acc[2][4];
    #pragma unroll
    for (int tm = 0; tm < 2; ++tm)
        #pragma unroll
        for (int tn = 0; tn < 4; ++tn)
            #pragma unroll
            for (int j = 0; j < 8; ++j)
                acc[tm][tn][j] = 0.0f;

    const int half = lane >> 4;   // 0/1
    const int rsub = lane & 15;   // row/col within 16

    auto compute_chunk = [&]() {
        // A fragments: contiguous k order (not AMD's table order): lane half selects K0 = half*16,
        // 16 sequential K (matches B)
        v16bf afh[2], afl[2];
        #pragma unroll
        for (int tm = 0; tm < 2; ++tm) {
            const __bf16* rph = sAhi + (wm * 32 + tm * 16 + rsub) * LDSW;
            const __bf16* rpl = sAlo + (wm * 32 + tm * 16 + rsub) * LDSW;
            ((uint4*)&afh[tm])[0] = *(const uint4*)(rph + half * 16);
            ((uint4*)&afh[tm])[1] = *(const uint4*)(rph + half * 16 + 8);
            ((uint4*)&afl[tm])[0] = *(const uint4*)(rpl + half * 16);
            ((uint4*)&afl[tm])[1] = *(const uint4*)(rpl + half * 16 + 8);
        }
        #pragma unroll
        for (int tn = 0; tn < 4; ++tn) {
            // B fragments: lane = column N, half selects K0 = half*16, 16 seq K
            const __bf16* rph = sBhi + (wn * 64 + tn * 16 + rsub) * LDSW + half * 16;
            const __bf16* rpl = sBlo + (wn * 64 + tn * 16 + rsub) * LDSW + half * 16;
            v16bf bfh, bfl;
            ((uint4*)&bfh)[0] = *(const uint4*)(rph);
            ((uint4*)&bfh)[1] = *(const uint4*)(rph + 8);
            ((uint4*)&bfl)[0] = *(const uint4*)(rpl);
            ((uint4*)&bfl)[1] = *(const uint4*)(rpl + 8);
            #pragma unroll
            for (int tm = 0; tm < 2; ++tm) {
                acc[tm][tn] = __builtin_amdgcn_wmma_f32_16x16x32_bf16(
                    false, afh[tm], false, bfh, (short)0, acc[tm][tn], false, false);
                acc[tm][tn] = __builtin_amdgcn_wmma_f32_16x16x32_bf16(
                    false, afl[tm], false, bfh, (short)0, acc[tm][tn], false, false);
                acc[tm][tn] = __builtin_amdgcn_wmma_f32_16x16x32_bf16(
                    false, afh[tm], false, bfl, (short)0, acc[tm][tn], false, false);
            }
        }
    };

    load_chunk(0);
    #pragma unroll 1
    for (int kc = 0; kc < KCH; ++kc) {
        __syncthreads();              // previous compute done reading LDS
        store_chunk(kc);
        __syncthreads();              // LDS tile ready
        if (kc + 1 < KCH) load_chunk(kc + 1);  // prefetch overlaps WMMAs
        compute_chunk();
    }

    // Epilogue: D layout (16x16 f32): VGPR j, lanes 0-15 -> M=j, N=lane;
    // lanes 16-31 -> M=8+j, N=lane-16.
    const float* av = avec + (size_t)bc * Lc;
    const float* bv = bvec + (size_t)bc * Lc;
    #pragma unroll
    for (int tm = 0; tm < 2; ++tm) {
        float avv[8];
        #pragma unroll
        for (int j = 0; j < 8; ++j)
            avv[j] = av[l0 + wm * 32 + tm * 16 + j + half * 8];
        #pragma unroll
        for (int tn = 0; tn < 4; ++tn) {
            const int mcol = m0 + wn * 64 + tn * 16 + rsub;
            const float bvv = bv[mcol];
            #pragma unroll
            for (int j = 0; j < 8; ++j) {
                const int lrow = l0 + wm * 32 + tm * 16 + j + half * 8;
                const float val = acc[tm][tn][j] + avv[j] + bvv;
                out[(((size_t)bb * Lc + lrow) * Lc + mcol) * Cc + cc] = val;
            }
        }
    }
}

// ---------------------------------------------------------------------------
extern "C" void kernel_launch(void* const* d_in, const int* in_sizes, int n_in,
                              void* d_out, int out_size, void* d_ws, size_t ws_size,
                              hipStream_t stream)
{
    const float* S = (const float*)d_in[0];   // start_hidden (B,C,L,H) f32
    const float* E = (const float*)d_in[1];   // end_hidden   (B,C,L,H) f32
    const float* v = (const float*)d_in[2];   // v (4H,) f32
    float* out = (float*)d_out;               // (B, L, L, C) f32

    float* avec = (float*)d_ws;                         // B*C*L floats
    float* bvec = avec + (size_t)Bc * Cc * Lc;          // B*C*L floats (128 KiB total)

    prep_rowdots<<<(Bc * Cc * Lc) / 8, 256, 0, stream>>>(S, E, v, avec, bvec);

    dim3 grid(Lc / BN, Lc / BM, Bc * Cc);
    gemm_bclm<<<grid, 256, 0, stream>>>(S, E, v, avec, bvec, out);
}
